// NoisyTopKGating_13245679141623
// MI455X (gfx1250) — matrix operand order repaired and checked by reading, not compiled, not run on hardware
//
#include <hip/hip_runtime.h>
#include <hip/hip_bf16.h>
#include <cmath>

typedef __attribute__((ext_vector_type(16))) __bf16 v16bf;
typedef __attribute__((ext_vector_type(8)))  float  v8f;

#define D_DIM 2048
#define E_DIM 64
#define KSTEPS (D_DIM / 32)        // 64 wmma K-steps
#define NTILES (E_DIM / 16)        // 4 N-tiles of 16
#define ROWS_TOTAL 16384           // B*T = 4*4096
#define WAVES_PER_BLOCK 4
#define ROWS_PER_WAVE 16
#define ROWS_PER_BLOCK (WAVES_PER_BLOCK * ROWS_PER_WAVE)
// packed layout: idx = ((((ks*4 + tile)*2 + mat)*32) + lane)*16 + j
// -> per (ks,tile,mat) fragment: 512 elems (1 KB); per ks: 4096 elems (8 KB)
#define PACK_ELEMS_TOTAL (KSTEPS * NTILES * 2 * 32 * 16)   // 262144 bf16 = 512 KB

// ---------------------------------------------------------------------------
// Pre-pass: convert Wg_w / Wn_w (f32, [E=64, D=2048] row-major) into bf16
// fragments laid out exactly as V_WMMA_F32_16X16X32_BF16's B operand
// (32x16, K-major per lane half), gate/noise interleaved per (ks,tile):
//   frag(ks,tile,mat): lane l, elem j -> W_mat[tile*16+(l&15)][ks*32+(j<8 ? 8*(l>>4)+j : 16+8*(l>>4)+j-8)]
// ---------------------------------------------------------------------------
__global__ void noisy_topk_pack_w(const float* __restrict__ Wg,
                                  const float* __restrict__ Wn,
                                  __bf16* __restrict__ pw) {
  int idx = blockIdx.x * blockDim.x + threadIdx.x;
  if (idx >= PACK_ELEMS_TOTAL) return;
  int j    = idx & 15;
  int lane = (idx >> 4) & 31;
  int mat  = (idx >> 9) & 1;
  int tile = (idx >> 10) & (NTILES - 1);
  int ks   = idx >> 12;
  int e = tile * 16 + (lane & 15);
  int k = ks * 32 + ((j < 8) ? (8 * (lane >> 4) + j) : (16 + 8 * (lane >> 4) + (j - 8)));
  const float* W = mat ? Wn : Wg;
  pw[idx] = (__bf16)W[e * D_DIM + k];
}

// ---------------------------------------------------------------------------
// Main kernel: per wave, 16 rows x all 64 experts, software-pipelined:
// iteration ks issues ALL loads for ks+1 (clause), then converts/computes ks,
// so the 8 WMMAs cover the next iteration's load latency.
// ---------------------------------------------------------------------------
__global__ __launch_bounds__(128) void noisy_topk_main(
    const float* __restrict__ x,      // [16384, 2048]
    const float* __restrict__ eps,    // [16384, 64]
    const float* __restrict__ Wg_b,   // [64]
    const float* __restrict__ Wn_b,   // [64]
    const __bf16* __restrict__ pw,    // packed weights (interleaved g/n)
    float* __restrict__ out) {        // [16384, 64]
  __shared__ float sh_h[WAVES_PER_BLOCK][ROWS_PER_WAVE][E_DIM];  // 16 KB
  __shared__ float sh_p1[WAVES_PER_BLOCK][ROWS_PER_WAVE];
  __shared__ float sh_p2[WAVES_PER_BLOCK][ROWS_PER_WAVE];
  __shared__ int   sh_i1[WAVES_PER_BLOCK][ROWS_PER_WAVE];
  __shared__ int   sh_i2[WAVES_PER_BLOCK][ROWS_PER_WAVE];

  const int lane = threadIdx.x & 31;
  const int wave = threadIdx.x >> 5;
  const int half = lane >> 4;       // 0: lanes 0-15, 1: lanes 16-31
  const int r    = lane & 15;

  const long wave_row_base = (long)blockIdx.x * ROWS_PER_BLOCK + (long)wave * ROWS_PER_WAVE;
  const long arow          = wave_row_base + r;   // A-matrix row for this lane

  v8f accG[NTILES], accN[NTILES];
#pragma unroll
  for (int t = 0; t < NTILES; ++t) { accG[t] = {}; accN[t] = {}; }

  const float* xrow  = x + arow * D_DIM;
  const __bf16* pwl  = pw + (size_t)lane * 16;    // per-lane fragment base

  // ---- pipeline prologue: issue loads for ks = 0 ----
  float4 araw[4];
  v16bf  bfr[8];
  {
    const float* xp = xrow;
    araw[0] = *(const float4*)(xp + 8 * half);
    araw[1] = *(const float4*)(xp + 8 * half + 4);
    araw[2] = *(const float4*)(xp + 16 + 8 * half);
    araw[3] = *(const float4*)(xp + 16 + 8 * half + 4);
    const __bf16* pl = pwl;                        // ks = 0
#pragma unroll
    for (int f = 0; f < 8; ++f) bfr[f] = *(const v16bf*)(pl + (size_t)f * 512);
  }

#pragma unroll 2
  for (int ks = 0; ks < KSTEPS; ++ks) {
    // ---- issue next iteration's loads first (clamped at the end) ----
    const int ksn = (ks + 1 < KSTEPS) ? ks + 1 : KSTEPS - 1;
    float4 araw_n[4];
    v16bf  bfr_n[8];
    {
      const float* xpn = xrow + ksn * 32;
      araw_n[0] = *(const float4*)(xpn + 8 * half);
      araw_n[1] = *(const float4*)(xpn + 8 * half + 4);
      araw_n[2] = *(const float4*)(xpn + 16 + 8 * half);
      araw_n[3] = *(const float4*)(xpn + 16 + 8 * half + 4);
      const __bf16* pln = pwl + (size_t)ksn * 4096;
#pragma unroll
      for (int f = 0; f < 8; ++f) bfr_n[f] = *(const v16bf*)(pln + (size_t)f * 512);
    }
    // keep HBM streaming ahead of the pipeline (~1 KB/row ahead)
    __builtin_prefetch(xrow + (ks + 8) * 32 + 8 * half, 0, 3);

    // ---- convert current A fragment f32 -> bf16 ----
    v16bf af;
    af[0]  = (__bf16)araw[0].x; af[1]  = (__bf16)araw[0].y;
    af[2]  = (__bf16)araw[0].z; af[3]  = (__bf16)araw[0].w;
    af[4]  = (__bf16)araw[1].x; af[5]  = (__bf16)araw[1].y;
    af[6]  = (__bf16)araw[1].z; af[7]  = (__bf16)araw[1].w;
    af[8]  = (__bf16)araw[2].x; af[9]  = (__bf16)araw[2].y;
    af[10] = (__bf16)araw[2].z; af[11] = (__bf16)araw[2].w;
    af[12] = (__bf16)araw[3].x; af[13] = (__bf16)araw[3].y;
    af[14] = (__bf16)araw[3].z; af[15] = (__bf16)araw[3].w;

    // ---- 8 WMMAs with the current (already-arrived) B fragments ----
#pragma unroll
    for (int t = 0; t < NTILES; ++t) {
      accG[t] = __builtin_amdgcn_wmma_f32_16x16x32_bf16(
          false, af, false, bfr[t * 2 + 0], (short)0, accG[t], false, false);
      accN[t] = __builtin_amdgcn_wmma_f32_16x16x32_bf16(
          false, af, false, bfr[t * 2 + 1], (short)0, accN[t], false, false);
    }

    // ---- rotate pipeline registers ----
#pragma unroll
    for (int q = 0; q < 4; ++q) araw[q] = araw_n[q];
#pragma unroll
    for (int f = 0; f < 8; ++f) bfr[f] = bfr_n[f];
  }

  // Epilogue: h = gate + b_g + eps * softplus(noise + b_n), into LDS.
  // softplus(z) = max(z,0) + log1p(exp(-|z|))  -- branch-free, stable.
  // C/D layout: VGPR v, lanes 0-15 -> row v (col = lane), lanes 16-31 -> row v+8.
#pragma unroll
  for (int t = 0; t < NTILES; ++t) {
    const int e   = t * 16 + r;
    const float bgb = Wg_b[e];
    const float bnb = Wn_b[e];
#pragma unroll
    for (int v = 0; v < 8; ++v) {
      const int  m  = v + 8 * half;
      const long rr = wave_row_base + m;
      const float ev = eps[rr * E_DIM + e];
      const float z  = accN[t][v] + bnb;
      const float sp = fmaxf(z, 0.0f) + log1pf(expf(-fabsf(z)));
      sh_h[wave][m][e] = accG[t][v] + bgb + ev * sp;
    }
  }
  __syncthreads();

  // Top-2 per row (lowest index wins ties, matching jax.lax.top_k), then
  // softmax over the two surviving logits.
  if (lane < ROWS_PER_WAVE) {
    float v1 = -3.402823466e38f, v2 = -3.402823466e38f;
    int   i1 = 0, i2 = 0;
    for (int e = 0; e < E_DIM; ++e) {
      const float v = sh_h[wave][lane][e];
      if (v > v1)      { v2 = v1; i2 = i1; v1 = v; i1 = e; }
      else if (v > v2) { v2 = v;  i2 = e; }
    }
    const float ed  = expf(v2 - v1);          // <= 1, safe
    const float inv = 1.0f / (1.0f + ed);
    sh_p1[wave][lane] = inv;
    sh_p2[wave][lane] = ed * inv;
    sh_i1[wave][lane] = i1;
    sh_i2[wave][lane] = i2;
  }
  __syncthreads();

  // Write output rows: zeros except the two winners.
#pragma unroll 4
  for (int m = 0; m < ROWS_PER_WAVE; ++m) {
    const long rr = wave_row_base + m;
    const int   i1 = sh_i1[wave][m], i2 = sh_i2[wave][m];
    const float p1 = sh_p1[wave][m], p2 = sh_p2[wave][m];
    const int e0 = lane;
    const int e1 = lane + 32;
    out[rr * E_DIM + e0] = (e0 == i1) ? p1 : ((e0 == i2) ? p2 : 0.0f);
    out[rr * E_DIM + e1] = (e1 == i1) ? p1 : ((e1 == i2) ? p2 : 0.0f);
  }
}

extern "C" void kernel_launch(void* const* d_in, const int* in_sizes, int n_in,
                              void* d_out, int out_size, void* d_ws, size_t ws_size,
                              hipStream_t stream) {
  (void)in_sizes; (void)n_in; (void)out_size; (void)ws_size;
  const float* x    = (const float*)d_in[0];
  const float* eps  = (const float*)d_in[1];
  const float* Wg_w = (const float*)d_in[2];
  const float* Wg_b = (const float*)d_in[3];
  const float* Wn_w = (const float*)d_in[4];
  const float* Wn_b = (const float*)d_in[5];
  float* out = (float*)d_out;

  // Workspace: one packed bf16 weight image (gate/noise interleaved), 512 KB.
  __bf16* pw = (__bf16*)d_ws;

  noisy_topk_pack_w<<<(PACK_ELEMS_TOTAL + 255) / 256, 256, 0, stream>>>(Wg_w, Wn_w, pw);
  noisy_topk_main<<<ROWS_TOTAL / ROWS_PER_BLOCK, 128, 0, stream>>>(
      x, eps, Wg_b, Wn_b, pw, out);
}
